// SAGE_82386062671991
// MI455X (gfx1250) — matrix operand order repaired and checked by reading, not compiled, not run on hardware
//
#include <hip/hip_runtime.h>
#include <hip/hip_bf16.h>

typedef __attribute__((ext_vector_type(16))) __bf16 v16bf;
typedef __attribute__((ext_vector_type(8)))  float  v8f;

#define NODES_PER_BLOCK 128   // 8 waves x 16-row tiles
#define D 128

// Native FP32 -> BF16 conversion (lowers to v_cvt_pk_bf16_f32 on gfx1250,
// replacing the 3-VALU-op manual RNE sequence per element).
static __device__ inline __bf16 f2bf(float f) {
    return static_cast<__bf16>(f);
}

// ---------------- utility: zero a float buffer (grid-stride) ----------------
__global__ void sage_zero_kernel(float* p, size_t n) {
    size_t i = (size_t)blockIdx.x * blockDim.x + threadIdx.x;
    size_t stride = (size_t)gridDim.x * blockDim.x;
    for (; i < n; i += stride) p[i] = 0.0f;
}

// ---------------- degree accumulation ----------------
__global__ void sage_degree_kernel(const int* __restrict__ dst, float* deg, int E) {
    int e = blockIdx.x * blockDim.x + threadIdx.x;
    if (e < E) atomicAdd(&deg[dst[e]], 1.0f);
}

// ---------------- scatter-add: agg[dst] += x[src], one wave per edge ----------------
__global__ void sage_scatter_kernel(const float* __restrict__ x,
                                    const int* __restrict__ src,
                                    const int* __restrict__ dst,
                                    float* agg, int E) {
    int gid  = blockIdx.x * blockDim.x + threadIdx.x;
    int e    = gid >> 5;
    int lane = gid & 31;
    if (e >= E) return;
    int s = src[e];
    int d = dst[e];
    const float4* xs = (const float4*)(x + (size_t)s * D);
    float4 v = xs[lane];                       // 32 lanes x float4 = 128 floats
    float* dp = agg + (size_t)d * D + lane * 4;
    atomicAdd(dp + 0, v.x);
    atomicAdd(dp + 1, v.y);
    atomicAdd(dp + 2, v.z);
    atomicAdd(dp + 3, v.w);
}

// ---------------- fused SAGE layer GEMM (bf16 WMMA, fp32 accumulate) ----------------
// out[i][:] = relu( (agg[i]/clamp(deg[i],1)) @ Wl^T + bias + self[i] @ Wr^T )
__global__ __launch_bounds__(256) void sage_gemm_kernel(
        const float* agg, const float* __restrict__ deg, const float* self,
        const float* __restrict__ Wl, const float* __restrict__ Wr,
        const float* __restrict__ bias, float* out, int nNodes, int doRelu) {
    __shared__ __bf16 sW[2][D * D];            // 64 KB LDS: bf16 Wl, Wr
    for (int i = threadIdx.x; i < D * D; i += 256) {
        sW[0][i] = f2bf(Wl[i]);
        sW[1][i] = f2bf(Wr[i]);
    }
    __syncthreads();

    const int wave = threadIdx.x >> 5;
    const int lane = threadIdx.x & 31;
    const int half = lane >> 4;                // 0: lanes 0-15, 1: lanes 16-31
    const int l16  = lane & 15;
    const int m0   = blockIdx.x * NODES_PER_BLOCK + wave * 16;
    int row  = m0 + l16;
    int rowc = row < nNodes ? row : nNodes - 1;
    float rdeg = 1.0f / fmaxf(deg[rowc], 1.0f);

    // Prefetch the self-row (phase 1) while phase 0 runs: global_prefetch_b8
    __builtin_prefetch(self + (size_t)rowc * D, 0, 3);
    __builtin_prefetch(self + (size_t)rowc * D + 64, 0, 3);

    v8f acc[8];
    #pragma unroll
    for (int t = 0; t < 8; ++t)
        #pragma unroll
        for (int i = 0; i < 8; ++i) acc[t][i] = 0.0f;

    #pragma unroll
    for (int phase = 0; phase < 2; ++phase) {
        const float* srcm = phase ? self : agg;
        const __bf16* w   = sW[phase];
        float scale = phase ? 1.0f : rdeg;
        const float* arow = srcm + (size_t)rowc * D;
        #pragma unroll
        for (int kk = 0; kk < D; kk += 32) {
            // A fragment (ISA 16-bit A 16x32 layout): lane-half selects K-octet
            int ka = kk + half * 8;
            const float4* a0 = (const float4*)(arow + ka);
            const float4* a1 = (const float4*)(arow + ka + 16);
            float4 p0 = a0[0], p1 = a0[1];
            float4 q0 = a1[0], q1 = a1[1];
            v16bf afrag;
            afrag[0]  = f2bf(p0.x * scale); afrag[1]  = f2bf(p0.y * scale);
            afrag[2]  = f2bf(p0.z * scale); afrag[3]  = f2bf(p0.w * scale);
            afrag[4]  = f2bf(p1.x * scale); afrag[5]  = f2bf(p1.y * scale);
            afrag[6]  = f2bf(p1.z * scale); afrag[7]  = f2bf(p1.w * scale);
            afrag[8]  = f2bf(q0.x * scale); afrag[9]  = f2bf(q0.y * scale);
            afrag[10] = f2bf(q0.z * scale); afrag[11] = f2bf(q0.w * scale);
            afrag[12] = f2bf(q1.x * scale); afrag[13] = f2bf(q1.y * scale);
            afrag[14] = f2bf(q1.z * scale); afrag[15] = f2bf(q1.w * scale);

            #pragma unroll
            for (int t = 0; t < 8; ++t) {
                // B fragment: lane = output channel n, lane-half selects K-octet
                int n = t * 16 + l16;
                const __bf16* wrow = w + n * D + kk + half * 8;
                v16bf bfrag;
                #pragma unroll
                for (int j = 0; j < 16; ++j) bfrag[j] = wrow[j < 8 ? j : j + 8];
                acc[t] = __builtin_amdgcn_wmma_f32_16x16x32_bf16(
                            false, afrag, false, bfrag, (short)0, acc[t],
                            false, false);
            }
        }
    }

    // Epilogue: C/D layout — element i holds row m0 + i + 8*half, col = l16 + 16*t
    #pragma unroll
    for (int t = 0; t < 8; ++t) {
        int n = t * 16 + l16;
        float b = bias[n];
        #pragma unroll
        for (int i = 0; i < 8; ++i) {
            int r = m0 + i + half * 8;
            if (r < nNodes) {
                float v = acc[t][i] + b;
                if (doRelu) v = fmaxf(v, 0.0f);
                out[(size_t)r * D + n] = v;
            }
        }
    }
}

// ---------------- layer 3: 128 -> 2, wave per node with shuffle reduction ----------------
__global__ void sage_out_kernel(const float* __restrict__ agg,
                                const float* __restrict__ deg,
                                const float* __restrict__ h,
                                const float* __restrict__ W3l,
                                const float* __restrict__ b3,
                                const float* __restrict__ W3r,
                                float* __restrict__ out, int nNodes) {
    int node = blockIdx.x * (blockDim.x >> 5) + (threadIdx.x >> 5);
    int lane = threadIdx.x & 31;
    if (node >= nNodes) return;
    float rdeg = 1.0f / fmaxf(deg[node], 1.0f);
    float s0 = 0.0f, s1 = 0.0f;
    const float* arow = agg + (size_t)node * D;
    const float* hrow = h   + (size_t)node * D;
    #pragma unroll
    for (int kb = 0; kb < D; kb += 32) {
        int k = kb + lane;
        float m  = arow[k] * rdeg;
        float hv = hrow[k];
        s0 += m * W3l[k]     + hv * W3r[k];
        s1 += m * W3l[D + k] + hv * W3r[D + k];
    }
    #pragma unroll
    for (int off = 16; off > 0; off >>= 1) {
        s0 += __shfl_down(s0, off);
        s1 += __shfl_down(s1, off);
    }
    if (lane == 0) {
        out[(size_t)node * 2 + 0] = s0 + b3[0];
        out[(size_t)node * 2 + 1] = s1 + b3[1];
    }
}

extern "C" void kernel_launch(void* const* d_in, const int* in_sizes, int n_in,
                              void* d_out, int out_size, void* d_ws, size_t ws_size,
                              hipStream_t stream) {
    const float* x   = (const float*)d_in[0];
    const int*   ei  = (const int*)d_in[1];
    const float* W1l = (const float*)d_in[2];
    const float* b1  = (const float*)d_in[3];
    const float* W1r = (const float*)d_in[4];
    const float* W2l = (const float*)d_in[5];
    const float* b2  = (const float*)d_in[6];
    const float* W2r = (const float*)d_in[7];
    const float* W3l = (const float*)d_in[8];
    const float* b3  = (const float*)d_in[9];
    const float* W3r = (const float*)d_in[10];

    const int N = in_sizes[0] / D;          // 100000
    const int E = in_sizes[1] / 2;          // 1600000
    const int* src = ei;
    const int* dst = ei + E;

    float* deg  = (float*)d_ws;             // N
    float* bufA = deg + N;                  // N*128  (agg scratch / h2)
    float* bufB = bufA + (size_t)N * D;     // N*128  (h1 / agg3)
    float* outp = (float*)d_out;            // N*2

    const int ZGRID   = 2048;
    const int DEGGRID = (E + 255) / 256;
    const int SCGRID  = (int)(((size_t)E * 32 + 255) / 256);
    const int GEMGRID = (N + NODES_PER_BLOCK - 1) / NODES_PER_BLOCK;
    const int OUTGRID = (N + 7) / 8;

    // degree (shared by all layers)
    sage_zero_kernel<<<ZGRID, 256, 0, stream>>>(deg, (size_t)N);
    sage_degree_kernel<<<DEGGRID, 256, 0, stream>>>(dst, deg, E);

    // ---- layer 1: h1 = relu(sage(x)) -> bufB ----
    sage_zero_kernel<<<ZGRID, 256, 0, stream>>>(bufA, (size_t)N * D);
    sage_scatter_kernel<<<SCGRID, 256, 0, stream>>>(x, src, dst, bufA, E);
    sage_gemm_kernel<<<GEMGRID, 256, 0, stream>>>(bufA, deg, x, W1l, W1r, b1,
                                                  bufB, N, 1);

    // ---- layer 2: h2 = relu(sage(h1)) -> bufA (in place over agg) ----
    sage_zero_kernel<<<ZGRID, 256, 0, stream>>>(bufA, (size_t)N * D);
    sage_scatter_kernel<<<SCGRID, 256, 0, stream>>>(bufB, src, dst, bufA, E);
    sage_gemm_kernel<<<GEMGRID, 256, 0, stream>>>(bufA, deg, bufB, W2l, W2r, b2,
                                                  bufA, N, 1);

    // ---- layer 3: out = sage(h2) (no relu), 128 -> 2 ----
    sage_zero_kernel<<<ZGRID, 256, 0, stream>>>(bufB, (size_t)N * D);
    sage_scatter_kernel<<<SCGRID, 256, 0, stream>>>(bufA, src, dst, bufB, E);
    sage_out_kernel<<<OUTGRID, 256, 0, stream>>>(bufB, deg, bufA, W3l, b3, W3r,
                                                 outp, N);
}
